// fixed_gru_53584011984942
// MI455X (gfx1250) — matrix operand order repaired and checked by reading, not compiled, not run on hardware
//
#include <hip/hip_runtime.h>
#include <hip/hip_bf16.h>
#include <math.h>

#define EMB   1024
#define STEPS 64
#define BATCH 256
#define BT    (BATCH * STEPS)   // 16384 rows of the batched x-projection GEMM
#define NX    3072              // [Xu | Xr | Xc]
#define NH    2048              // [W_hu | W_hr] concatenated along N

typedef __attribute__((ext_vector_type(16))) __bf16         v16bf;
typedef __attribute__((ext_vector_type(8)))  float          v8f;
typedef __attribute__((ext_vector_type(8)))  unsigned short v8us;
typedef __attribute__((ext_vector_type(16))) unsigned short v16us;

// ---------------- workspace layout (bytes) ----------------
static constexpr size_t SZ_TXTB = (size_t)BT * EMB * 2;        // txt in bf16
static constexpr size_t SZ_WX   = (size_t)NX * EMB * 2;        // [W_iu;W_ir;W_i] bf16
static constexpr size_t SZ_WH   = (size_t)NH * EMB * 2;        // [W_hu;W_hr] bf16
static constexpr size_t SZ_WHH  = (size_t)EMB * EMB * 2;       // W_h bf16
static constexpr size_t SZ_P    = (size_t)BT * NX * 4;         // x-projections fp32
static constexpr size_t SZ_H    = (size_t)BATCH * EMB * 4;     // h fp32
static constexpr size_t SZ_HB   = (size_t)BATCH * EMB * 2;     // h bf16
static constexpr size_t SZ_U    = (size_t)BATCH * EMB * 4;     // update gate fp32
static constexpr size_t SZ_RHB  = (size_t)BATCH * EMB * 2;     // (r*h) bf16

static constexpr size_t OFF_TXTB = 0;
static constexpr size_t OFF_WX   = OFF_TXTB + SZ_TXTB;
static constexpr size_t OFF_WH   = OFF_WX + SZ_WX;
static constexpr size_t OFF_WHH  = OFF_WH + SZ_WH;
static constexpr size_t OFF_P    = OFF_WHH + SZ_WHH;
static constexpr size_t OFF_H    = OFF_P + SZ_P;
static constexpr size_t OFF_HB   = OFF_H + SZ_H;
static constexpr size_t OFF_U    = OFF_HB + SZ_HB;
static constexpr size_t OFF_RHB  = OFF_U + SZ_U;

// ---------------- helpers ----------------
__device__ __forceinline__ unsigned short f2bf(float f) {
  unsigned int u = __float_as_uint(f);
  u += 0x7FFFu + ((u >> 16) & 1u);      // round-to-nearest-even
  return (unsigned short)(u >> 16);
}
__device__ __forceinline__ float sigmoidf_(float x) {
  return 1.0f / (1.0f + __expf(-x));
}

// Wave-tiled WMMA GEMM core:  acc[MT][NT] += A[BM x K] * (Bw[BN x K])^T
// A, Bw row-major bf16 (Bw holds B transposed, i.e. W[N][K]).
// Fragment layouts per CDNA5 ISA 7.12.2 (16-bit A 16x32, B 32x16, f32 C/D 16x16).
template<int BM, int BN, int WAVES_M, int WAVES_N, int MT, int NT>
__device__ __forceinline__ void wmma_gemm_core(
    const unsigned short* __restrict__ A,
    const unsigned short* __restrict__ Bw,
    int K, int m_block, int n_block,
    v8f acc[MT][NT], int& wm0, int& wn0, int& lane_out)
{
  const int tid  = threadIdx.x;
  const int lane = tid & 31;
  const int wave = tid >> 5;
  const int wM   = wave % WAVES_M;
  const int wN   = wave / WAVES_M;
  wm0 = m_block * BM + wM * (MT * 16);
  wn0 = n_block * BN + wN * (NT * 16);
  lane_out = lane;
  const int half = lane >> 4;
  const int l15  = lane & 15;

#pragma unroll
  for (int mt = 0; mt < MT; ++mt)
#pragma unroll
    for (int nt = 0; nt < NT; ++nt) { v8f z = {}; acc[mt][nt] = z; }

  for (int kc = 0; kc < K; kc += 32) {
    v16bf afrag[MT];
#pragma unroll
    for (int mt = 0; mt < MT; ++mt) {
      // A frag: lane holds row (wm0+mt*16+l15); lanes 0-15 K={kc..kc+7, kc+16..kc+23},
      // lanes 16-31 K={kc+8..kc+15, kc+24..kc+31}  -> two contiguous 16B loads.
      const unsigned short* pa = A + (size_t)(wm0 + mt * 16 + l15) * K + kc + half * 8;
      v8us lo = *(const v8us*)pa;
      v8us hi = *(const v8us*)(pa + 16);
      v16us t;
#pragma unroll
      for (int i = 0; i < 8; ++i) { t[i] = lo[i]; t[i + 8] = hi[i]; }
      afrag[mt] = __builtin_bit_cast(v16bf, t);
    }
#pragma unroll
    for (int nt = 0; nt < NT; ++nt) {
      // B frag: lane holds column n=(wn0+nt*16+l15); lanes 0-15 K={kc..kc+7, kc+16..kc+23},
      // lanes 16-31 K={kc+8..kc+15, kc+24..kc+31} -> two contiguous 16B loads from row-major W[N][K].
      const unsigned short* pb = Bw + (size_t)(wn0 + nt * 16 + l15) * K + kc + half * 8;
      v8us blo = *(const v8us*)pb;
      v8us bhi = *(const v8us*)(pb + 16);
      v16us tb;
#pragma unroll
      for (int i = 0; i < 8; ++i) { tb[i] = blo[i]; tb[i + 8] = bhi[i]; }
      v16bf bfrag = __builtin_bit_cast(v16bf, tb);
#pragma unroll
      for (int mt = 0; mt < MT; ++mt)
        acc[mt][nt] = __builtin_amdgcn_wmma_f32_16x16x32_bf16(
            false, afrag[mt], false, bfrag, (short)0, acc[mt][nt], false, false);
    }
  }
}

// ---------------- kernels ----------------

// fp32 -> bf16 (row-major copy)
__global__ void cvt_f32_bf16_k(const float* __restrict__ src,
                               unsigned short* __restrict__ dst, int n) {
  int i = blockIdx.x * blockDim.x + threadIdx.x;
  if (i < n) dst[i] = f2bf(src[i]);
}

// P[BT x 3072] = txt_bf16 @ [W_iu|W_ir|W_i]^T + [b_iu|b_ir|b_i]
__global__ __launch_bounds__(256) void xproj_k(
    const unsigned short* __restrict__ Xb, const unsigned short* __restrict__ Wx,
    const float* __restrict__ b_iu, const float* __restrict__ b_ir,
    const float* __restrict__ b_i, float* __restrict__ P)
{
  constexpr int BM = 128, BN = 128, WVM = 4, WVN = 2, MT = 2, NT = 4;
  v8f acc[MT][NT]; int wm0, wn0, lane;
  wmma_gemm_core<BM, BN, WVM, WVN, MT, NT>(Xb, Wx, EMB, blockIdx.y, blockIdx.x,
                                           acc, wm0, wn0, lane);
  const int half = lane >> 4, l15 = lane & 15;
#pragma unroll
  for (int mt = 0; mt < MT; ++mt)
#pragma unroll
    for (int nt = 0; nt < NT; ++nt) {
      const int n = wn0 + nt * 16 + l15;
      const float bias = (n < EMB) ? b_iu[n]
                       : (n < 2 * EMB ? b_ir[n - EMB] : b_i[n - 2 * EMB]);
#pragma unroll
      for (int j = 0; j < 8; ++j) {
        const int m = wm0 + mt * 16 + half * 8 + j;
        P[(size_t)m * NX + n] = acc[mt][nt][j] + bias;
      }
    }
}

// h0 = tanh(Xc[:,0,:]) ; seed h (fp32 + bf16) and out[:,0,:]
__global__ void init_h_k(const float* __restrict__ P, float* __restrict__ out,
                         float* __restrict__ H, unsigned short* __restrict__ Hb)
{
  int i = blockIdx.x * blockDim.x + threadIdx.x;   // 0 .. BATCH*EMB-1
  int m = i >> 10, n = i & (EMB - 1);
  float h = tanhf(P[(size_t)(m * STEPS) * NX + 2 * EMB + n]);
  out[(size_t)(m * STEPS) * EMB + n] = h;
  H[i] = h;
  Hb[i] = f2bf(h);
}

// Phase A (per step): [u|r] = sigmoid(h @ [W_hu|W_hr]^T + b + X); emit u, (r*h)_bf16
// 64x64 tiles -> 128 blocks: maximize in-flight blocks during the latency-bound
// serial section (weights are L2-resident on the 192MB L2).
__global__ __launch_bounds__(128) void gates_k(
    const unsigned short* __restrict__ Hb, const unsigned short* __restrict__ Wh,
    const float* __restrict__ b_hu, const float* __restrict__ b_hr,
    const float* __restrict__ P, const float* __restrict__ H,
    float* __restrict__ U, unsigned short* __restrict__ RHb, int t)
{
  constexpr int BM = 64, BN = 64, WVM = 2, WVN = 2, MT = 2, NT = 2;
  v8f acc[MT][NT]; int wm0, wn0, lane;
  wmma_gemm_core<BM, BN, WVM, WVN, MT, NT>(Hb, Wh, EMB, blockIdx.y, blockIdx.x,
                                           acc, wm0, wn0, lane);
  const int half = lane >> 4, l15 = lane & 15;
#pragma unroll
  for (int mt = 0; mt < MT; ++mt)
#pragma unroll
    for (int nt = 0; nt < NT; ++nt) {
      const int n = wn0 + nt * 16 + l15;
#pragma unroll
      for (int j = 0; j < 8; ++j) {
        const int m = wm0 + mt * 16 + half * 8 + j;   // batch row
        const float pre = acc[mt][nt][j] + P[(size_t)(m * STEPS + t) * NX + n];
        if (n < EMB) {
          U[m * EMB + n] = sigmoidf_(pre + b_hu[n]);
        } else {
          const int nn = n - EMB;
          const float r = sigmoidf_(pre + b_hr[nn]);
          RHb[m * EMB + nn] = f2bf(r * H[m * EMB + nn]);
        }
      }
    }
}

// Phase B (per step): c = tanh((r*h) @ W_h^T + b_h + Xc); h = u*h + (1-u)*c
__global__ __launch_bounds__(128) void cand_k(
    const unsigned short* __restrict__ RHb, const unsigned short* __restrict__ Whh,
    const float* __restrict__ b_h, const float* __restrict__ P,
    const float* __restrict__ U, float* __restrict__ H,
    unsigned short* __restrict__ Hb, float* __restrict__ out, int t)
{
  constexpr int BM = 64, BN = 64, WVM = 2, WVN = 2, MT = 2, NT = 2;
  v8f acc[MT][NT]; int wm0, wn0, lane;
  wmma_gemm_core<BM, BN, WVM, WVN, MT, NT>(RHb, Whh, EMB, blockIdx.y, blockIdx.x,
                                           acc, wm0, wn0, lane);
  const int half = lane >> 4, l15 = lane & 15;
#pragma unroll
  for (int mt = 0; mt < MT; ++mt)
#pragma unroll
    for (int nt = 0; nt < NT; ++nt) {
      const int n = wn0 + nt * 16 + l15;
      const float bias = b_h[n];
#pragma unroll
      for (int j = 0; j < 8; ++j) {
        const int m = wm0 + mt * 16 + half * 8 + j;   // batch row
        const float c = tanhf(acc[mt][nt][j] + bias +
                              P[(size_t)(m * STEPS + t) * NX + 2 * EMB + n]);
        const float u = U[m * EMB + n];
        const float hold = H[m * EMB + n];
        const float hn = u * hold + (1.0f - u) * c;
        out[(size_t)(m * STEPS + t) * EMB + n] = hn;
        H[m * EMB + n]  = hn;
        Hb[m * EMB + n] = f2bf(hn);
      }
    }
}

// ---------------- host launcher ----------------
extern "C" void kernel_launch(void* const* d_in, const int* in_sizes, int n_in,
                              void* d_out, int out_size, void* d_ws, size_t ws_size,
                              hipStream_t stream)
{
  (void)in_sizes; (void)n_in; (void)out_size; (void)ws_size;
  const float* txt  = (const float*)d_in[0];
  const float* W_iu = (const float*)d_in[1];
  const float* b_iu = (const float*)d_in[2];
  const float* W_hu = (const float*)d_in[3];
  const float* b_hu = (const float*)d_in[4];
  const float* W_ir = (const float*)d_in[5];
  const float* b_ir = (const float*)d_in[6];
  const float* W_hr = (const float*)d_in[7];
  const float* b_hr = (const float*)d_in[8];
  const float* W_i  = (const float*)d_in[9];
  const float* b_i  = (const float*)d_in[10];
  const float* W_h  = (const float*)d_in[11];
  const float* b_h  = (const float*)d_in[12];
  float* out = (float*)d_out;

  char* ws = (char*)d_ws;
  unsigned short* txtb = (unsigned short*)(ws + OFF_TXTB);
  unsigned short* Wx   = (unsigned short*)(ws + OFF_WX);
  unsigned short* Wh   = (unsigned short*)(ws + OFF_WH);
  unsigned short* Whh  = (unsigned short*)(ws + OFF_WHH);
  float*          P    = (float*)(ws + OFF_P);
  float*          H    = (float*)(ws + OFF_H);
  unsigned short* Hb   = (unsigned short*)(ws + OFF_HB);
  float*          U    = (float*)(ws + OFF_U);
  unsigned short* RHb  = (unsigned short*)(ws + OFF_RHB);

  const int WEL = EMB * EMB;  // elements per weight matrix

  // fp32 -> bf16 staging (weights concatenated along N; txt flattened [B*T, D])
  cvt_f32_bf16_k<<<(BT * EMB) / 256, 256, 0, stream>>>(txt, txtb, BT * EMB);
  cvt_f32_bf16_k<<<WEL / 256, 256, 0, stream>>>(W_iu, Wx,            WEL);
  cvt_f32_bf16_k<<<WEL / 256, 256, 0, stream>>>(W_ir, Wx + WEL,      WEL);
  cvt_f32_bf16_k<<<WEL / 256, 256, 0, stream>>>(W_i,  Wx + 2 * WEL,  WEL);
  cvt_f32_bf16_k<<<WEL / 256, 256, 0, stream>>>(W_hu, Wh,            WEL);
  cvt_f32_bf16_k<<<WEL / 256, 256, 0, stream>>>(W_hr, Wh + WEL,      WEL);
  cvt_f32_bf16_k<<<WEL / 256, 256, 0, stream>>>(W_h,  Whh,           WEL);

  // Batched input projections for all (b,t):  P = txt @ [W_iu|W_ir|W_i]^T + bias
  xproj_k<<<dim3(NX / 128, BT / 128), 256, 0, stream>>>(txtb, Wx, b_iu, b_ir, b_i, P);

  // t = 0: h0 = tanh(Xc)
  init_h_k<<<(BATCH * EMB) / 256, 256, 0, stream>>>(P, out, H, Hb);

  // Sequential scan: two dependent WMMA GEMMs per step, weights L2-resident.
  // 64x64 tiles -> 128 / 64 blocks per launch for latency hiding.
  for (int t = 1; t < STEPS; ++t) {
    gates_k<<<dim3(NH / 64, BATCH / 64), 128, 0, stream>>>(
        Hb, Wh, b_hu, b_hr, P, H, U, RHb, t);
    cand_k<<<dim3(EMB / 64, BATCH / 64), 128, 0, stream>>>(
        RHb, Whh, b_h, P, U, H, Hb, out, t);
  }
}
